// NonMaxSuppression_60911226192176
// MI455X (gfx1250) — matrix operand order repaired and checked by reading, not compiled, not run on hardware
//
#include <hip/hip_runtime.h>

// NMS for (B=8, N=2048, 6) predictions -> (boxes[8,100,4], scores[8,100],
// classes[8,100], num_det[8]) concatenated flat as float into d_out (4808 el).
//
// One workgroup per image, fully LDS-resident:
//  - bitonic sort of (score,idx) u64 keys; wave-local stages (jj<=16) skip the
//    workgroup barrier (wave32 LDS ops are in-order within a wave); branchless
//    compare-exchange
//  - suppression bitmask built 64 rows at a time; class equality computed
//    exactly with ONE v_wmma_f32_16x16x32_f16 per 16x16 tile using a split
//    one-hot encoding: class c = 8q+r, K[0..9] one-hot of q, K[24..31]
//    one-hot of r  =>  dot = [q1==q2]+[r1==r2], same-class <=> dot > 1.5
//  - IoU test without division; branchless predicate feeding wave32 ballots
//  - per-tile bit write-out: 8 ballots redistributed via a cndmask select
//    tree to lanes 0..15, ONE ds_store_b16 per tile (no atomics, no zeroing)
//  - wave-synchronous greedy suppression on a 2048-bit keep mask
//  - per-class cap (<=100) + first-100 emission, single-thread O(N) pass

#define NMS_CLASSES 80
#define NMS_CONF_T 0.05f
#define NMS_IOU_T 0.5f
#define NMS_MAX_DET 100
#define NMS_MAX_PER_CLASS 100
#define NMS_N 2048
#define NMS_B 8

typedef __attribute__((ext_vector_type(16))) _Float16 v16h;
typedef __attribute__((ext_vector_type(8)))  float    v8f;

__device__ __forceinline__ unsigned class_mask(int c) {
  // split one-hot: bit q (0..9) for c>>3, bit 24+r for c&7
  return (1u << (c >> 3)) | (1u << (24 + (c & 7)));
}

__global__ __launch_bounds__(1024)
void nms_kernel(const float* __restrict__ pred, float* __restrict__ out) {
  const int b    = blockIdx.x;
  const int tid  = threadIdx.x;
  const int lane = tid & 31;
  const int wave = tid >> 5;

  // 16 KB sort keys aliased with 16 KB suppression block.
  // supb16[row][tn] (64 x 128 u16) == supb[row][w] (64 x 64 u32) bit-for-bit:
  // u32 word w = supb16[2w] | supb16[2w+1]<<16  -> j-bits [w*32, w*32+32).
  __shared__ union SharedU {
    unsigned long long skv[NMS_N];
    unsigned           supb[64 * 64];
    unsigned short     supb16[64 * 128];
  } u;
  __shared__ float          sbox[NMS_N * 4];   // sorted boxes (16B, aligned)
  __shared__ float          sscore[NMS_N];     // sorted scores
  __shared__ unsigned char  sclsb[NMS_N];      // sorted classes (0..79)
  __shared__ unsigned       keepw[64];         // 2048-bit keep mask
  __shared__ unsigned short ord[NMS_MAX_DET];  // output order (sorted indices)
  __shared__ unsigned       ccnt[NMS_CLASSES]; // per-class counters
  __shared__ int            ndet_s;

  const size_t base = (size_t)b * NMS_N;

  // ---- Phase 1: build sort keys (score bits | idx); invalid -> key 0 ------
  for (int rep = 0; rep < 2; ++rep) {
    int j = tid + rep * 1024;
    const float* rp = pred + (base + j) * 6;
    __builtin_prefetch(rp, 0, 0);  // global_prefetch_b8: warm L2 for gather
    float s = rp[5];
    unsigned kb = (s > NMS_CONF_T) ? (__float_as_uint(s) | 0x80000000u) : 0u;
    u.skv[j] = ((unsigned long long)kb << 32) | (unsigned)j;
  }
  __syncthreads();

  // ---- Phase 2: bitonic sort, descending (ties: larger idx first) ---------
  // jj>=32: partners cross waves -> s_barrier.  jj<=16: each 2*jj window is
  // owned by one wave (wave32 LDS is in-order per wave) -> no hw barrier.
  {
    volatile unsigned long long* vskv = u.skv;
    for (unsigned kk = 2; kk <= NMS_N; kk <<= 1) {
      for (unsigned jj = kk >> 1; jj > 0; jj >>= 1) {
        if (jj >= 32) __syncthreads();
        else          __builtin_amdgcn_wave_barrier();
        unsigned t = (unsigned)tid;
        unsigned a = (t / jj) * (jj << 1) + (t % jj);
        unsigned c = a + jj;
        unsigned long long x = vskv[a], y = vskv[c];
        bool up = ((a & kk) == 0);   // descending segment
        unsigned long long mx = (x > y) ? x : y;
        unsigned long long mn = (x > y) ? y : x;
        vskv[a] = up ? mx : mn;      // branchless: always store both
        vskv[c] = up ? mn : mx;
      }
    }
  }
  __syncthreads();

  // ---- Phase 3: gather sorted boxes / classes / scores into LDS -----------
  for (int rep = 0; rep < 2; ++rep) {
    int j = tid + rep * 1024;
    unsigned idx = (unsigned)(u.skv[j] & 0xFFFFFFFFu);
    const float* rp = pred + (base + idx) * 6;
    sbox[j * 4 + 0] = rp[0];
    sbox[j * 4 + 1] = rp[1];
    sbox[j * 4 + 2] = rp[2];
    sbox[j * 4 + 3] = rp[3];
    sclsb[j] = (unsigned char)(int)rp[4];
    sscore[j] = rp[5];
  }
  __syncthreads();

  // ---- keep mask init: validity in sorted order (ballot per wave) ---------
  for (int rep = 0; rep < 2; ++rep) {
    int j = tid + rep * 1024;  // j>>5 == rep*32 + wave, bit == lane
    unsigned bal = __builtin_amdgcn_ballot_w32(sscore[j] > NMS_CONF_T);
    if (lane == 0) keepw[rep * 32 + wave] = bal;
  }
  __syncthreads();

  // ---- Main loop: 32 blocks of 64 rows ------------------------------------
  const int tm  = wave & 3;   // row-tile within the 64-row block (4 tiles)
  const int tnb = wave >> 2;  // base column-tile (0..7), step 8 -> 128 tiles
  const int nA  = lane & 15;
  const int hiA = (lane & 16) ? 8 : 0;    // A: K-offset; C/D: M-offset
  const int hiB = (lane & 16) ? 8 : 0;    // B: K-offset
  volatile unsigned* vk = keepw;
  const float4* sbox4 = (const float4*)sbox;

  for (int step = 0; step < NMS_N / 64; ++step) {
    const int i0 = step * 64;

    // A fragment (16x32 f16). A layout: lanes L and L+16 both hold row
    // M=L&15; element e maps to K = e + (e&8 ? 8 : 0) + hiA.
    const unsigned mA = class_mask(sclsb[i0 + tm * 16 + nA]) >> hiA;
    v16h Af{};
#pragma unroll
    for (int e = 0; e < 16; ++e) {
      const int sh = e + ((e & 8) ? 8 : 0);  // compile-time constant
      Af[e] = (_Float16)(((mA >> sh) & 1u) ? 1.0f : 0.0f);
    }

    // Row boxes for this wave's 8 C-rows (invariant across column tiles).
    // C layout: VGPR r, lane<16 -> M=r; lane>=16 -> M=r+8; N=lane&15.
    float rx1[8], ry1[8], rx2[8], ry2[8], rar[8];
#pragma unroll
    for (int r = 0; r < 8; ++r) {
      const int gi = i0 + tm * 16 + r + hiA;
      const float4 rb = sbox4[gi];           // one ds_load_b128
      rx1[r] = rb.x; ry1[r] = rb.y; rx2[r] = rb.z; ry2[r] = rb.w;
      rar[r] = fmaxf(rb.z - rb.x, 0.0f) * fmaxf(rb.w - rb.y, 0.0f);
    }

    for (int t = 0; t < 16; ++t) {
      const int tn   = tnb + (t << 3);
      const int coln = tn * 16 + nA;

      // B fragment (32x16 f16): e maps to K = e + (e&8 ? 8 : 0) + hiB
      const unsigned mB = class_mask(sclsb[coln]) >> hiB;
      v16h Bf{};
#pragma unroll
      for (int e = 0; e < 16; ++e) {
        Bf[e] = (_Float16)(((mB >> (e + ((e & 8) ? 8 : 0))) & 1u) ? 1.0f : 0.0f);
      }

      // dot = [q_row==q_col] + [r_row==r_col]; same class <=> dot == 2
      v8f c = {};
      c = __builtin_amdgcn_wmma_f32_16x16x32_f16(false, Af, false, Bf,
                                                 (short)0, c, false, false);

      // column box (one ds_load_b128), shared by all 8 rows this lane covers
      const float4 cb = sbox4[coln];
      const float car = fmaxf(cb.z - cb.x, 0.0f) * fmaxf(cb.w - cb.y, 0.0f);

      unsigned bal[8];
#pragma unroll
      for (int r = 0; r < 8; ++r) {
        const int gi = i0 + tm * 16 + r + hiA;
        const float ix1 = fmaxf(rx1[r], cb.x);
        const float iy1 = fmaxf(ry1[r], cb.y);
        const float ix2 = fminf(rx2[r], cb.z);
        const float iy2 = fminf(ry2[r], cb.w);
        const float inter = fmaxf(ix2 - ix1, 0.0f) * fmaxf(iy2 - iy1, 0.0f);
        const float uni = rar[r] + car - inter;
        // Branchless: iou > IOU_T <=> inter > IOU_T * max(uni, 1e-8)
        const bool bit = (bool)((c[r] > 1.5f) &
                                (inter > NMS_IOU_T * fmaxf(uni, 1e-8f)) &
                                (coln > gi));
        // ballot: bits[0..15] = row (tm*16+r), bits[16..31] = row (+8)
        bal[r] = __builtin_amdgcn_ballot_w32(bit);
      }

      // Redistribute 8 uniform ballots to lanes 0..15 (row = tm*16 + lane):
      // lane<8 takes bal[lane] low half, lane>=8 takes bal[lane-8] high half.
      const unsigned s01 = (lane & 1) ? bal[1] : bal[0];
      const unsigned s23 = (lane & 1) ? bal[3] : bal[2];
      const unsigned s45 = (lane & 1) ? bal[5] : bal[4];
      const unsigned s67 = (lane & 1) ? bal[7] : bal[6];
      const unsigned s03 = (lane & 2) ? s23 : s01;
      const unsigned s47 = (lane & 2) ? s67 : s45;
      const unsigned sel = (lane & 4) ? s47 : s03;
      const unsigned hlf = (lane & 8) ? (sel >> 16) : (sel & 0xFFFFu);
      if (lane < 16) {
        // each (row, tn) u16 cell written exactly once -> no atomics/zeroing
        u.supb16[(tm * 16 + lane) * 128 + tn] = (unsigned short)hlf;
      }
    }
    __syncthreads();

    // Sequential greedy suppression for rows [i0, i0+64), wave 0 only.
    // LDS ops within a wave are in-order; volatile prevents caching.
    if (tid < 32) {
      for (int ii = 0; ii < 64; ++ii) {
        const int i = i0 + ii;
        const unsigned kb = vk[i >> 5];
        if ((kb >> (i & 31)) & 1u) {
          vk[tid]      &= ~u.supb[(ii << 6) + tid];
          vk[tid + 32] &= ~u.supb[(ii << 6) + tid + 32];
        }
      }
    }
    __syncthreads();
  }

  // ---- per-class cap + first-100 selection --------------------------------
  if (tid < NMS_CLASSES) ccnt[tid] = 0u;
  __syncthreads();
  if (tid == 0) {
    int r2 = 0;
    for (int j = 0; j < NMS_N; ++j) {
      const unsigned w2 = keepw[j >> 5];
      if ((w2 >> (j & 31)) & 1u) {
        const int cc = sclsb[j];
        const unsigned cnt = ++ccnt[cc];
        if (cnt > NMS_MAX_PER_CLASS) {
          keepw[j >> 5] = w2 & ~(1u << (j & 31));
        } else {
          if (r2 < NMS_MAX_DET) ord[r2] = (unsigned short)j;
          ++r2;
        }
      }
    }
    ndet_s = (r2 < NMS_MAX_DET) ? r2 : NMS_MAX_DET;
  }
  __syncthreads();

  // ---- emit ---------------------------------------------------------------
  if (tid < NMS_MAX_DET) {
    const int s = tid;
    float bx0 = 0.f, bx1 = 0.f, bx2 = 0.f, bx3 = 0.f, sc = 0.f, cf = 0.f;
    if (s < ndet_s) {
      const int j = ord[s];
      bx0 = sbox[j * 4 + 0];
      bx1 = sbox[j * 4 + 1];
      bx2 = sbox[j * 4 + 2];
      bx3 = sbox[j * 4 + 3];
      sc = sscore[j];
      cf = (float)sclsb[j];
    }
    float* bo = out + ((size_t)b * NMS_MAX_DET + s) * 4;
    bo[0] = bx0; bo[1] = bx1; bo[2] = bx2; bo[3] = bx3;
    out[NMS_B * NMS_MAX_DET * 4 + b * NMS_MAX_DET + s] = sc;
    out[NMS_B * NMS_MAX_DET * 5 + b * NMS_MAX_DET + s] = cf;
  }
  if (tid == 0) {
    out[NMS_B * NMS_MAX_DET * 6 + b] = (float)ndet_s;  // num_det as float
  }
}

extern "C" void kernel_launch(void* const* d_in, const int* in_sizes, int n_in,
                              void* d_out, int out_size, void* d_ws,
                              size_t ws_size, hipStream_t stream) {
  (void)in_sizes; (void)n_in; (void)d_ws; (void)ws_size; (void)out_size;
  const float* pred = (const float*)d_in[0];
  float* out = (float*)d_out;
  nms_kernel<<<NMS_B, 1024, 0, stream>>>(pred, out);
}
